// Attention_Head_59219009077952
// MI455X (gfx1250) — matrix operand order repaired and checked by reading, not compiled, not run on hardware
//
#include <hip/hip_runtime.h>

// ---- types --------------------------------------------------------------
typedef __bf16 bf16;
typedef __attribute__((ext_vector_type(16))) __bf16 v16bf;
typedef __attribute__((ext_vector_type(8)))  float  v8f;

#define B_  8
#define T_  2048
#define E_  1024
#define D_  128

// =========================================================================
// Kernel 1: QKV projection  out = x @ W + b
//   grid.x = (B*T)/128 row tiles, grid.y = {0:q, 1:k, 2:v}
//   256 threads = 8 waves; wave w owns rows [w*16, w*16+16), full N=128.
//   q,k stored row-major bf16 [m][128]; v stored transposed bf16 [b][d][t].
// =========================================================================
__global__ void __launch_bounds__(256)
qkv_gemm_kernel(const float* __restrict__ x,
                const float* __restrict__ Wq, const float* __restrict__ bq,
                const float* __restrict__ Wk, const float* __restrict__ bk,
                const float* __restrict__ Wv, const float* __restrict__ bv,
                bf16* __restrict__ qbuf, bf16* __restrict__ kbuf,
                bf16* __restrict__ vT)
{
    const int which = blockIdx.y;
    const float* __restrict__ W    = (which == 0) ? Wq : (which == 1) ? Wk : Wv;
    const float* __restrict__ bias = (which == 0) ? bq : (which == 1) ? bk : bv;

    const int m0   = blockIdx.x * 128;
    const int tid  = threadIdx.x;
    const int wave = tid >> 5;
    const int lane = tid & 31;
    const int r    = lane & 15;
    const int hi   = lane >> 4;

    // LDS tiles: row stride 40 halves = 80B (16B-aligned rows)
    __shared__ __align__(16) bf16 At[128 * 40];   // x tile   [m][k], 128x32
    __shared__ __align__(16) bf16 Bt[128 * 40];   // W^T tile [n][k], 128x32

    v8f acc[8] = {};

    for (int k0 = 0; k0 < E_; k0 += 32) {
        // stage x tile (f32 -> bf16): thread -> row=tid>>1, 16 cols
        {
            const int row = tid >> 1, c0 = (tid & 1) * 16;
            const float4* src =
                (const float4*)&x[(size_t)(m0 + row) * E_ + k0 + c0];
            // prefetch next K-slice of this row (global_prefetch_b8)
            if (k0 + 32 < E_)
                __builtin_prefetch(&x[(size_t)(m0 + row) * E_ + k0 + 32 + c0], 0, 3);
            bf16* dst = &At[row * 40 + c0];
            #pragma unroll
            for (int v = 0; v < 4; ++v) {
                float4 f = src[v];
                dst[v * 4 + 0] = (bf16)f.x; dst[v * 4 + 1] = (bf16)f.y;
                dst[v * 4 + 2] = (bf16)f.z; dst[v * 4 + 3] = (bf16)f.w;
            }
        }
        // stage W^T tile: thread -> kk=tid>>3, 16 n-cols, scatter transpose
        {
            const int kk = tid >> 3, n0 = (tid & 7) * 16;
            const float4* src =
                (const float4*)&W[(size_t)(k0 + kk) * D_ + n0];
            #pragma unroll
            for (int v = 0; v < 4; ++v) {
                float4 f = src[v];
                Bt[(n0 + v * 4 + 0) * 40 + kk] = (bf16)f.x;
                Bt[(n0 + v * 4 + 1) * 40 + kk] = (bf16)f.y;
                Bt[(n0 + v * 4 + 2) * 40 + kk] = (bf16)f.z;
                Bt[(n0 + v * 4 + 3) * 40 + kk] = (bf16)f.w;
            }
        }
        __syncthreads();

        // A fragment (16x32 bf16): lane(r,hi) -> row r, K range hi*16 .. hi*16+15
        v16bf a;
        {
            const bf16* arow = &At[(wave * 16 + r) * 40];
            #pragma unroll
            for (int j = 0; j < 8; ++j) {
                a[j]     = arow[hi * 16 + j];
                a[8 + j] = arow[hi * 16 + 8 + j];
            }
        }
        // batch-load ALL B fragments, then run 8 back-to-back WMMAs
        v16bf bfr[8];
        #pragma unroll
        for (int nt = 0; nt < 8; ++nt) {
            const bf16* brow = &Bt[(nt * 16 + r) * 40 + hi * 16];
            #pragma unroll
            for (int j = 0; j < 16; ++j) bfr[nt][j] = brow[j];
        }
        #pragma unroll
        for (int nt = 0; nt < 8; ++nt)
            acc[nt] = __builtin_amdgcn_wmma_f32_16x16x32_bf16(
                false, a, false, bfr[nt], (short)0, acc[nt], false, false);
        __syncthreads();
    }

    // epilogue: bias add, store (C layout: vgpr j -> row j + 8*hi, col r)
    #pragma unroll
    for (int nt = 0; nt < 8; ++nt) {
        const int   col = nt * 16 + r;
        const float bv_ = bias[col];
        #pragma unroll
        for (int j = 0; j < 8; ++j) {
            const float o = acc[nt][j] + bv_;
            const int   m = m0 + wave * 16 + j + hi * 8;
            if (which == 0) {
                qbuf[(size_t)m * D_ + col] = (bf16)o;
            } else if (which == 1) {
                kbuf[(size_t)m * D_ + col] = (bf16)o;
            } else {
                const int b = m >> 11, t = m & (T_ - 1);
                vT[((size_t)b * D_ + col) * T_ + t] = (bf16)o;
            }
        }
    }
}

// =========================================================================
// Kernel 2: flash attention  out = softmax(q k^T / sqrt(D)) v
//   grid.x = T/128 query tiles, grid.y = batch
//   256 threads = 8 waves; wave w owns 16 query rows. Keys in tiles of 64,
//   double-buffered in LDS via GLOBAL_LOAD_ASYNC_TO_LDS_B128 (ASYNCcnt).
// =========================================================================
#define KT_ 64   // keys per tile

__global__ void __launch_bounds__(256)
flash_attn_kernel(const bf16* __restrict__ qbuf, const bf16* __restrict__ kbuf,
                  const bf16* __restrict__ vT, float* __restrict__ out)
{
    const int b    = blockIdx.y;
    const int q0   = blockIdx.x * 128;
    const int tid  = threadIdx.x;
    const int wave = tid >> 5;
    const int lane = tid & 31;
    const int r    = lane & 15;
    const int hi   = lane >> 4;

    __shared__ __align__(16) bf16 kT[2][64 * 136];   // K tile   [key][d]
    __shared__ __align__(16) bf16 vTl[2][128 * 72];  // V^T tile [d][key]
    __shared__ __align__(16) bf16 pLds[8 * 16 * 72]; // per-wave P patch

    // issue async global->LDS copies for key tile starting at kt into buf
    auto issue_tile = [&](int kt, int buf) {
        #pragma unroll
        for (int it = 0; it < 4; ++it) {
            const int c = tid * 4 + it;           // 1024 16B-chunks per tile
            {   // kT: 64 rows x 128 halves, 16 chunks/row
                const int row = c >> 4, c8 = (c & 15) * 8;
                unsigned dst = (unsigned)(size_t)&kT[buf][row * 136 + c8];
                const bf16* src = &kbuf[((size_t)b * T_ + kt + row) * D_ + c8];
                asm volatile("global_load_async_to_lds_b128 %0, %1, off"
                             :: "v"(dst), "v"(src) : "memory");
            }
            {   // vTl: 128 rows x 64 halves, 8 chunks/row
                const int d = c >> 3, c8 = (c & 7) * 8;
                unsigned dst = (unsigned)(size_t)&vTl[buf][d * 72 + c8];
                const bf16* src = &vT[((size_t)b * D_ + d) * T_ + kt + c8];
                asm volatile("global_load_async_to_lds_b128 %0, %1, off"
                             :: "v"(dst), "v"(src) : "memory");
            }
        }
    };

    // q fragments (A operand, reused across all key tiles): 4 d-slices of 32
    v16bf qf[4];
    {
        const bf16* qrow = &qbuf[((size_t)b * T_ + q0 + wave * 16 + r) * D_];
        #pragma unroll
        for (int ds = 0; ds < 4; ++ds)
            #pragma unroll
            for (int j = 0; j < 8; ++j) {
                qf[ds][j]     = qrow[ds * 32 + hi * 16 + j];
                qf[ds][8 + j] = qrow[ds * 32 + hi * 16 + 8 + j];
            }
    }

    v8f   acc[8] = {};
    float mI[8], lI[8];
    #pragma unroll
    for (int j = 0; j < 8; ++j) { mI[j] = -1e30f; lI[j] = 0.0f; }
    const float scale = 0.08838834764831845f;  // D^-0.5

    issue_tile(0, 0);

    for (int i = 0; i < T_ / KT_; ++i) {
        const int buf = i & 1;
        asm volatile("s_wait_asynccnt 0x0" ::: "memory");
        __syncthreads();                       // tile i resident, buf^1 free
        if (i + 1 < T_ / KT_) issue_tile((i + 1) * KT_, buf ^ 1);

        // S = q k^T : per d-slice, batch-load 4 K fragments then 4 WMMAs
        v8f s[4] = {};
        #pragma unroll
        for (int ds = 0; ds < 4; ++ds) {
            v16bf kf[4];
            #pragma unroll
            for (int nt = 0; nt < 4; ++nt) {
                const bf16* kr = &kT[buf][(nt * 16 + r) * 136 + ds * 32 + hi * 16];
                #pragma unroll
                for (int j = 0; j < 16; ++j) kf[nt][j] = kr[j];
            }
            #pragma unroll
            for (int nt = 0; nt < 4; ++nt)
                s[nt] = __builtin_amdgcn_wmma_f32_16x16x32_bf16(
                    false, qf[ds], false, kf[nt], (short)0, s[nt], false, false);
        }

        // online softmax; row j+8*hi lives across the lane's 16-lane half
        #pragma unroll
        for (int j = 0; j < 8; ++j) {
            const float a0 = s[0][j] * scale;
            const float a1 = s[1][j] * scale;
            const float a2 = s[2][j] * scale;
            const float a3 = s[3][j] * scale;
            float mx = fmaxf(fmaxf(a0, a1), fmaxf(a2, a3));
            #pragma unroll
            for (int m = 1; m <= 8; m <<= 1)
                mx = fmaxf(mx, __shfl_xor(mx, m, 32));
            const float mNew  = fmaxf(mI[j], mx);
            const float alpha = __expf(mI[j] - mNew);
            const float p0 = __expf(a0 - mNew);
            const float p1 = __expf(a1 - mNew);
            const float p2 = __expf(a2 - mNew);
            const float p3 = __expf(a3 - mNew);
            float rs = (p0 + p1) + (p2 + p3);
            #pragma unroll
            for (int m = 1; m <= 8; m <<= 1)
                rs += __shfl_xor(rs, m, 32);
            lI[j] = lI[j] * alpha + rs;
            mI[j] = mNew;
            #pragma unroll
            for (int n = 0; n < 8; ++n) acc[n][j] *= alpha;
            // C-layout -> A-layout transpose via per-wave LDS patch
            bf16* prow = &pLds[(wave * 16 + j + hi * 8) * 72];
            prow[r]      = (bf16)p0;
            prow[16 + r] = (bf16)p1;
            prow[32 + r] = (bf16)p2;
            prow[48 + r] = (bf16)p3;
        }

        // reload P as two A operands (same-wave LDS ops are in-order)
        v16bf pf0, pf1;
        {
            const bf16* prow = &pLds[(wave * 16 + r) * 72];
            #pragma unroll
            for (int j = 0; j < 8; ++j) {
                pf0[j]     = prow[hi * 16 + j];
                pf0[8 + j] = prow[hi * 16 + 8 + j];
                pf1[j]     = prow[32 + hi * 16 + j];
                pf1[8 + j] = prow[40 + hi * 16 + j];
            }
        }
        // acc += P @ V : process d-tiles in pairs, 4 frag loads then 4 WMMAs
        #pragma unroll
        for (int n0 = 0; n0 < 8; n0 += 2) {
            v16bf vf[4];
            #pragma unroll
            for (int p = 0; p < 2; ++p) {
                const bf16* v0r = &vTl[buf][((n0 + p) * 16 + r) * 72 + hi * 16];
                #pragma unroll
                for (int j = 0; j < 16; ++j) {
                    vf[2 * p + 0][j] = v0r[j];
                    vf[2 * p + 1][j] = v0r[32 + j];
                }
            }
            #pragma unroll
            for (int p = 0; p < 2; ++p) {
                acc[n0 + p] = __builtin_amdgcn_wmma_f32_16x16x32_bf16(
                    false, pf0, false, vf[2 * p + 0], (short)0, acc[n0 + p], false, false);
                acc[n0 + p] = __builtin_amdgcn_wmma_f32_16x16x32_bf16(
                    false, pf1, false, vf[2 * p + 1], (short)0, acc[n0 + p], false, false);
            }
        }
    }

    // epilogue: normalize and store f32
    #pragma unroll
    for (int j = 0; j < 8; ++j) {
        const float inv = 1.0f / lI[j];
        const int   row = q0 + wave * 16 + j + hi * 8;
        float* orow = &out[((size_t)b * T_ + row) * D_];
        #pragma unroll
        for (int n = 0; n < 8; ++n) orow[n * 16 + r] = acc[n][j] * inv;
    }
}

// =========================================================================
extern "C" void kernel_launch(void* const* d_in, const int* in_sizes, int n_in,
                              void* d_out, int out_size, void* d_ws, size_t ws_size,
                              hipStream_t stream) {
    const float* x  = (const float*)d_in[0];
    const float* Wq = (const float*)d_in[1];
    const float* bq = (const float*)d_in[2];
    const float* Wk = (const float*)d_in[3];
    const float* bk = (const float*)d_in[4];
    const float* Wv = (const float*)d_in[5];
    const float* bv = (const float*)d_in[6];
    float* out = (float*)d_out;

    const size_t MN = (size_t)B_ * T_ * D_;       // 16384*128
    bf16* qbuf = (bf16*)d_ws;                     // [B*T][D]   4 MB
    bf16* kbuf = qbuf + MN;                       // [B*T][D]   4 MB
    bf16* vT   = kbuf + MN;                       // [B][D][T]  4 MB

    qkv_gemm_kernel<<<dim3((B_ * T_) / 128, 3), 256, 0, stream>>>(
        x, Wq, bq, Wk, bk, Wv, bv, qbuf, kbuf, vT);
    flash_attn_kernel<<<dim3(T_ / 128, B_), 256, 0, stream>>>(
        qbuf, kbuf, vT, out);
}
